// GroupedQueryAttention_59983513256626
// MI455X (gfx1250) — matrix operand order repaired and checked by reading, not compiled, not run on hardware
//
#include <hip/hip_runtime.h>
#include <math.h>

typedef __attribute__((ext_vector_type(16))) __bf16 v16bf16;
typedef __attribute__((ext_vector_type(8)))  float  v8f;
typedef __attribute__((ext_vector_type(4)))  unsigned int u32x4;
typedef __attribute__((ext_vector_type(4)))  int i32x4;
typedef __attribute__((ext_vector_type(8)))  int i32x8;

#define D_MODEL 2048
#define KV_DIM  512
#define HEADS   32
#define SEQ     2048
#define BATCH   2

__device__ __forceinline__ unsigned short f2bf(float f) {
    union { float f; unsigned int u; } x; x.f = f;
    unsigned int u = x.u;
    u += 0x7fffu + ((u >> 16) & 1u);   // round-to-nearest-even
    return (unsigned short)(u >> 16);
}

__global__ void cvt_f32_bf16(const float* __restrict__ in,
                             unsigned short* __restrict__ out, int n) {
    int i = blockIdx.x * blockDim.x + threadIdx.x;
    if (i < n) out[i] = f2bf(in[i]);
}

union AFrag { v16bf16 v; uint4 u[2]; };

// ---------------------------------------------------------------------------
// Tensor Data Mover: 2D bf16 tile load Global -> LDS with row padding.
// Builds the D# descriptor per CDNA5 ISA 8.3/8.4 (groups 2/3 zero => 2D).
// Issue from ONE wave only; follow with s_wait_tensorcnt(0) + block barrier.
// pad_interval_code: pad every 2^(c+1) DWORDs; pad_amount_code: (a+1) DWORDs.
// NOTE: this toolchain exposes the 6-arg builtin (clang-23/therock form).
// ---------------------------------------------------------------------------
__device__ __forceinline__ void tdm_load_2d_bf16(
    const unsigned short* gptr, unsigned lds_byte_addr,
    unsigned tile_d0, unsigned tile_d1,            // elements (d0 = row length)
    unsigned long long row_stride_elems,
    unsigned pad_interval_code, unsigned pad_amount_code)
{
    unsigned long long ga = (unsigned long long)(uintptr_t)gptr;
    u32x4 g0 = { 1u,                                             // count=1 (user)
                 lds_byte_addr,                                  // lds_addr
                 (unsigned)(ga & 0xffffffffull),                 // global_addr lo
                 (unsigned)((ga >> 32) & 0x01ffffffull)          // global_addr[56:32]
                     | (2u << 30) };                             // type = 2 (image)
    i32x8 g1 = {0,0,0,0,0,0,0,0};
    g1[0] = (int)((1u << 16) |                                   // data_size = 2B
                  (1u << 20) |                                   // pad_enable
                  (pad_interval_code << 22) |
                  (pad_amount_code << 25));
    // tensor_dim0/1 := tile dims (tiles are always fully in-bounds here)
    g1[1] = (int)((tile_d0 & 0xffffu) << 16);                    // dim0[15:0]
    g1[2] = (int)((tile_d0 >> 16) | ((tile_d1 & 0xffffu) << 16));// dim0[31:16], dim1[15:0]
    g1[3] = (int)((tile_d1 >> 16) | ((tile_d0 & 0xffffu) << 16));// dim1[31:16], tile_dim0
    g1[4] = (int)(tile_d1 & 0xffffu);                            // tile_dim1 (tile_dim2=0)
    g1[5] = (int)(row_stride_elems & 0xffffffffull);             // dim0_stride[31:0]
    g1[6] = (int)((row_stride_elems >> 32) & 0xffffull);         // dim0_stride[47:32]
    i32x4 z4 = {0,0,0,0};
    i32x8 z8 = {0,0,0,0,0,0,0,0};
    __builtin_amdgcn_tensor_load_to_lds(g0, g1, z4, z4, z8, 0);
}

// ---------------------------------------------------------------------------
// Tiled bf16 GEMM:  C[M,N] = A[M,K] * W[K,N] + bias.   BM=128 BN=64 BK=32.
// 256 threads = 8 waves in a 4(M) x 2(N) grid; each wave -> 32x32 tile.
// A tile staged by the Tensor Data Mover (padded 32->40 elems/row);
// W tile transposed into LDS by the waves while the TDM runs.
// ---------------------------------------------------------------------------
template <int OUTF32>
__global__ __launch_bounds__(256) void gemm_bf16(
    const unsigned short* __restrict__ A,   // [M,K] bf16 row-major
    const unsigned short* __restrict__ W,   // [K,N] bf16 row-major
    const float* __restrict__ bias,         // [N]
    void* __restrict__ Cout,                // bf16 or f32 [M,N]
    int M, int N, int K)
{
    constexpr int BM = 128, BN = 64, BK = 32, LDA = 40, LDB = 40;
    __shared__ __attribute__((aligned(16))) unsigned short As[BM * LDA];
    __shared__ __attribute__((aligned(16))) unsigned short Bs[BN * LDB]; // [n][k]

    const int tid  = threadIdx.x;
    const int wave = tid >> 5, lane = tid & 31;
    const int hf   = lane >> 4, l16 = lane & 15;
    const int wm   = wave >> 1, wn  = wave & 1;
    const int m0   = blockIdx.x * BM;
    const int n0   = blockIdx.y * BN;
    const unsigned As_lds = (unsigned)(uintptr_t)&As[0];   // LDS aperture: addr[31:0]

    v8f acc[2][2];
#pragma unroll
    for (int i = 0; i < 2; i++)
#pragma unroll
        for (int j = 0; j < 2; j++) acc[i][j] = (v8f){0,0,0,0,0,0,0,0};

    for (int k0 = 0; k0 < K; k0 += BK) {
        __syncthreads();
        if (tid < 32) {   // wave 0: DMA the A tile (128 rows x 32 k, pad to 40)
            tdm_load_2d_bf16(A + (size_t)m0 * K + k0, As_lds,
                             /*tile_d0=*/BK, /*tile_d1=*/BM,
                             (unsigned long long)K,
                             /*pad_interval 16 DW=*/3u, /*pad 4 DW=*/3u);
        }
        {   // all waves: W tile (32 k x 64 n) transposed into Bs[n][k]
            int kk = tid >> 3;
            int n8 = (tid & 7) * 8;
            uint4 w = *(const uint4*)(W + (size_t)(k0 + kk) * N + n0 + n8);
            if (k0 + BK < K)
                __builtin_prefetch(W + (size_t)(k0 + BK + kk) * N + n0 + n8, 0, 3);
            const unsigned short* ws = (const unsigned short*)&w;
#pragma unroll
            for (int i = 0; i < 8; i++) Bs[(n8 + i) * LDB + kk] = ws[i];
        }
        if (tid < 32) __builtin_amdgcn_s_wait_tensorcnt(0);
        __syncthreads();

        AFrag af[2], bfr[2];
#pragma unroll
        for (int tm = 0; tm < 2; tm++) {
            const unsigned short* p = As + (wm * 32 + tm * 16 + l16) * LDA;
            af[tm].u[0] = *(const uint4*)(p + 8 * hf);
            af[tm].u[1] = *(const uint4*)(p + 16 + 8 * hf);
        }
#pragma unroll
        for (int tn = 0; tn < 2; tn++) {
            const unsigned short* p = Bs + (wn * 32 + tn * 16 + l16) * LDB + 8 * hf;
            bfr[tn].u[0] = *(const uint4*)(p);
            bfr[tn].u[1] = *(const uint4*)(p + 16);
        }
#pragma unroll
        for (int tm = 0; tm < 2; tm++)
#pragma unroll
            for (int tn = 0; tn < 2; tn++)
                acc[tm][tn] = __builtin_amdgcn_wmma_f32_16x16x32_bf16(
                    false, af[tm].v, false, bfr[tn].v,
                    (short)0, acc[tm][tn], false, false);
    }

#pragma unroll
    for (int tm = 0; tm < 2; tm++)
#pragma unroll
        for (int tn = 0; tn < 2; tn++) {
            int col = n0 + wn * 32 + tn * 16 + l16;
            float bv = bias[col];
#pragma unroll
            for (int r = 0; r < 8; r++) {
                int row = m0 + wm * 32 + tm * 16 + r + 8 * hf;
                float v = acc[tm][tn][r] + bv;
                if (OUTF32) ((float*)Cout)[(size_t)row * N + col] = v;
                else ((unsigned short*)Cout)[(size_t)row * N + col] = f2bf(v);
            }
        }
}

// ---------------------------------------------------------------------------
// Flash GQA attention. grid = (S/128, HEADS, BATCH), 256 threads = 8 waves.
// Wave w owns 16 query rows; KV streamed in 64-key blocks through LDS.
// K tile staged by TDM (row copy, pad 64->72); V transposed by the waves.
// ---------------------------------------------------------------------------
__global__ __launch_bounds__(256) void gqa_attn(
    const unsigned short* __restrict__ Q,   // [B*S, 2048] bf16
    const unsigned short* __restrict__ Km,  // [B*S, 512]  bf16
    const unsigned short* __restrict__ Vm,  // [B*S, 512]  bf16
    unsigned short* __restrict__ O)         // [B*S, 2048] bf16
{
    constexpr int BQ = 128, BKV = 64, LD = 72;
    __shared__ __attribute__((aligned(16))) unsigned short Ks[BKV * LD];      // [key][dh]
    __shared__ __attribute__((aligned(16))) unsigned short Vs[64 * LD];       // [dh][key]
    __shared__ __attribute__((aligned(16))) unsigned short Ps[8 * 16 * LD];   // per-wave P

    const int tid  = threadIdx.x;
    const int wave = tid >> 5, lane = tid & 31;
    const int hf   = lane >> 4, l16 = lane & 15;
    const int h    = blockIdx.y, b = blockIdx.z;
    const int kvh  = h >> 2;                      // repeat_interleave mapping
    const size_t tokbase = (size_t)b * SEQ;
    const int qrow0 = blockIdx.x * BQ + wave * 16;
    const unsigned Ks_lds = (unsigned)(uintptr_t)&Ks[0];

    // Q fragments for this wave's 16 rows (Dh=64 -> 2 k-fragments), loaded once.
    AFrag qf[2];
    {
        const unsigned short* qp =
            Q + (tokbase + qrow0 + l16) * D_MODEL + h * 64;
#pragma unroll
        for (int fr = 0; fr < 2; fr++) {
            qf[fr].u[0] = *(const uint4*)(qp + fr * 32 + 8 * hf);
            qf[fr].u[1] = *(const uint4*)(qp + fr * 32 + 16 + 8 * hf);
        }
    }

    v8f   oacc[4];
    float mrow[8], lrow[8];
#pragma unroll
    for (int d = 0; d < 4; d++) oacc[d] = (v8f){0,0,0,0,0,0,0,0};
#pragma unroll
    for (int r = 0; r < 8; r++) { mrow[r] = -INFINITY; lrow[r] = 0.0f; }
    const float scale = 0.125f;   // 1/sqrt(64)

    for (int j = 0; j < SEQ; j += BKV) {
        __syncthreads();
        if (tid < 32) {   // wave 0: DMA K block [64 keys x 64 dh] (== WMMA B^T layout)
            tdm_load_2d_bf16(Km + (tokbase + j) * KV_DIM + kvh * 64, Ks_lds,
                             /*tile_d0=*/64, /*tile_d1=*/BKV,
                             (unsigned long long)KV_DIM,
                             /*pad_interval 32 DW=*/4u, /*pad 4 DW=*/3u);
        }
        {   // all waves: stage V block transposed -> Vs[dh][key]
            int key = tid >> 2, dho = (tid & 3) * 16;
            const uint4* src = (const uint4*)(Vm + (tokbase + j + key) * KV_DIM + kvh * 64 + dho);
            uint4 v0 = src[0], v1 = src[1];
            if (j + BKV < SEQ)
                __builtin_prefetch(Vm + (tokbase + j + BKV + key) * KV_DIM + kvh * 64 + dho, 0, 3);
            const unsigned short* vs = (const unsigned short*)&v0;
#pragma unroll
            for (int i = 0; i < 8; i++) Vs[(dho + i) * LD + key] = vs[i];
            vs = (const unsigned short*)&v1;
#pragma unroll
            for (int i = 0; i < 8; i++) Vs[(dho + 8 + i) * LD + key] = vs[i];
        }
        if (tid < 32) __builtin_amdgcn_s_wait_tensorcnt(0);
        __syncthreads();

        // scores: 16 x 64 = 4 C-tiles, each accumulated over Dh with 2 WMMAs
        float sc[4][8];
#pragma unroll
        for (int t = 0; t < 4; t++) {
            AFrag b0, b1;
            const unsigned short* p = Ks + (t * 16 + l16) * LD;
            b0.u[0] = *(const uint4*)(p + 8 * hf);
            b0.u[1] = *(const uint4*)(p + 8 * hf + 16);
            b1.u[0] = *(const uint4*)(p + 32 + 8 * hf);
            b1.u[1] = *(const uint4*)(p + 32 + 8 * hf + 16);
            v8f c = (v8f){0,0,0,0,0,0,0,0};
            c = __builtin_amdgcn_wmma_f32_16x16x32_bf16(false, qf[0].v, false, b0.v,
                                                        (short)0, c, false, false);
            c = __builtin_amdgcn_wmma_f32_16x16x32_bf16(false, qf[1].v, false, b1.v,
                                                        (short)0, c, false, false);
#pragma unroll
            for (int r = 0; r < 8; r++) sc[t][r] = c[r] * scale;
        }

        // online softmax (rows live across 16-lane halves in C layout)
        float corr[8];
#pragma unroll
        for (int r = 0; r < 8; r++) {
            float bm = fmaxf(fmaxf(sc[0][r], sc[1][r]), fmaxf(sc[2][r], sc[3][r]));
#pragma unroll
            for (int msk = 1; msk < 16; msk <<= 1)
                bm = fmaxf(bm, __shfl_xor(bm, msk, 32));
            float nm = fmaxf(mrow[r], bm);
            corr[r]  = __expf(mrow[r] - nm);
            mrow[r]  = nm;
        }
        float rs[8];
#pragma unroll
        for (int r = 0; r < 8; r++) {
            float s = 0.0f;
#pragma unroll
            for (int t = 0; t < 4; t++) {
                float p = __expf(sc[t][r] - mrow[r]);
                sc[t][r] = p;
                s += p;
            }
#pragma unroll
            for (int msk = 1; msk < 16; msk <<= 1)
                s += __shfl_xor(s, msk, 32);
            rs[r] = s;
        }
#pragma unroll
        for (int r = 0; r < 8; r++) lrow[r] = lrow[r] * corr[r] + rs[r];
#pragma unroll
        for (int d = 0; d < 4; d++)
#pragma unroll
            for (int r = 0; r < 8; r++) oacc[d][r] *= corr[r];

        // P: C layout -> bf16 A layout via per-wave LDS staging
        unsigned short* pw = Ps + wave * 16 * LD;
#pragma unroll
        for (int t = 0; t < 4; t++)
#pragma unroll
            for (int r = 0; r < 8; r++)
                pw[(r + 8 * hf) * LD + t * 16 + l16] = f2bf(sc[t][r]);
        __syncthreads();

        AFrag pf[2];
        {
            const unsigned short* pp = pw + l16 * LD;
#pragma unroll
            for (int fr = 0; fr < 2; fr++) {
                pf[fr].u[0] = *(const uint4*)(pp + 8 * hf + 32 * fr);
                pf[fr].u[1] = *(const uint4*)(pp + 16 + 8 * hf + 32 * fr);
            }
        }

        // O += P @ V : 4 dh-tiles, 2 key-fragments each
#pragma unroll
        for (int d = 0; d < 4; d++) {
            AFrag v0, v1;
            const unsigned short* vp = Vs + (d * 16 + l16) * LD;
            v0.u[0] = *(const uint4*)(vp + 8 * hf);
            v0.u[1] = *(const uint4*)(vp + 8 * hf + 16);
            v1.u[0] = *(const uint4*)(vp + 32 + 8 * hf);
            v1.u[1] = *(const uint4*)(vp + 32 + 8 * hf + 16);
            oacc[d] = __builtin_amdgcn_wmma_f32_16x16x32_bf16(false, pf[0].v, false, v0.v,
                                                              (short)0, oacc[d], false, false);
            oacc[d] = __builtin_amdgcn_wmma_f32_16x16x32_bf16(false, pf[1].v, false, v1.v,
                                                              (short)0, oacc[d], false, false);
        }
    }

    // normalize and store to token-major [B*S, 2048] bf16
    float inv[8];
#pragma unroll
    for (int r = 0; r < 8; r++) inv[r] = 1.0f / lrow[r];
#pragma unroll
    for (int d = 0; d < 4; d++)
#pragma unroll
        for (int r = 0; r < 8; r++) {
            int row = qrow0 + r + 8 * hf;
            int col = h * 64 + d * 16 + l16;
            O[(tokbase + row) * D_MODEL + col] = f2bf(oacc[d][r] * inv[r]);
        }
}

// ---------------------------------------------------------------------------
extern "C" void kernel_launch(void* const* d_in, const int* in_sizes, int n_in,
                              void* d_out, int out_size, void* d_ws, size_t ws_size,
                              hipStream_t stream) {
    const float* x  = (const float*)d_in[0];
    const float* Wq = (const float*)d_in[1];
    const float* bq = (const float*)d_in[2];
    const float* Wk = (const float*)d_in[3];
    const float* bk = (const float*)d_in[4];
    const float* Wv = (const float*)d_in[5];
    const float* bv = (const float*)d_in[6];
    const float* Wo = (const float*)d_in[7];
    const float* bo = (const float*)d_in[8];

    const int M = BATCH * SEQ;           // 4096 tokens
    char* ws = (char*)d_ws;
    size_t off = 0;
    auto alloc = [&](size_t elems) {
        unsigned short* p = (unsigned short*)(ws + off);
        off += elems * 2; off = (off + 255) & ~(size_t)255;
        return p;
    };
    unsigned short* Xb  = alloc((size_t)M * D_MODEL);
    unsigned short* Wqb = alloc((size_t)D_MODEL * D_MODEL);
    unsigned short* Wkb = alloc((size_t)D_MODEL * KV_DIM);
    unsigned short* Wvb = alloc((size_t)D_MODEL * KV_DIM);
    unsigned short* Wob = alloc((size_t)D_MODEL * D_MODEL);
    unsigned short* Qb  = alloc((size_t)M * D_MODEL);
    unsigned short* Kb  = alloc((size_t)M * KV_DIM);
    unsigned short* Vb  = alloc((size_t)M * KV_DIM);
    unsigned short* Ob  = alloc((size_t)M * D_MODEL);

    auto cvt = [&](const float* s, unsigned short* d, int n) {
        cvt_f32_bf16<<<(n + 255) / 256, 256, 0, stream>>>(s, d, n);
    };
    cvt(x,  Xb,  M * D_MODEL);
    cvt(Wq, Wqb, D_MODEL * D_MODEL);
    cvt(Wk, Wkb, D_MODEL * KV_DIM);
    cvt(Wv, Wvb, D_MODEL * KV_DIM);
    cvt(Wo, Wob, D_MODEL * D_MODEL);

    dim3 blk(256);
    gemm_bf16<0><<<dim3(M / 128, D_MODEL / 64), blk, 0, stream>>>(
        Xb, Wqb, bq, (void*)Qb, M, D_MODEL, D_MODEL);
    gemm_bf16<0><<<dim3(M / 128, KV_DIM / 64), blk, 0, stream>>>(
        Xb, Wkb, bk, (void*)Kb, M, KV_DIM, D_MODEL);
    gemm_bf16<0><<<dim3(M / 128, KV_DIM / 64), blk, 0, stream>>>(
        Xb, Wvb, bv, (void*)Vb, M, KV_DIM, D_MODEL);

    gqa_attn<<<dim3(SEQ / 128, HEADS, BATCH), blk, 0, stream>>>(Qb, Kb, Vb, Ob);

    gemm_bf16<1><<<dim3(M / 128, D_MODEL / 64), blk, 0, stream>>>(
        Ob, Wob, bo, d_out, M, D_MODEL, D_MODEL);
}
